// wModel2_27058293965080
// MI455X (gfx1250) — compile-verified, matrix operand order checked by reading, not run on hardware
//
#include <hip/hip_runtime.h>
#include <hip/hip_bf16.h>

// MI455X / gfx1250 implementation.
// One workgroup (256 thr = 8 wave32) per batch sample; all per-sample matrices
// live in LDS; every GEMM runs on v_wmma_f32_16x16x32_f16 tiles with f32
// accumulation. Adjacency is compacted to a bitmask in LDS (pair read once).
// Weight tiles move global->LDS via ASYNCcnt-tracked async copies, overlapped
// with staging/compute. The 4-head conv is algebraically pre-reduced into one
// band-GEMM P = X^T @ wsum.

typedef __attribute__((ext_vector_type(16))) _Float16 v16h;
typedef __attribute__((ext_vector_type(8)))  _Float16 v8h;
typedef __attribute__((ext_vector_type(4)))  _Float16 v4h;
typedef __attribute__((ext_vector_type(8)))  float    v8f;
typedef _Float16 h16;

#define LRA   0.2f
#define EPSBN 1e-5f
#define NEGV  -9e15f

#define BATCH 2048
#define NN    101     // nodes
#define NP    112     // nodes padded to 16
#define KD    84      // Kmer feature dim
#define KDP   96
#define VD    92      // vec feature dim
#define VL    101
#define F     30      // hidden
#define FP    32
#define NH    4
#define HF    120
#define HFP   128     // also K-pad for attention GEMMs (101 -> 128)
#define TOUT  72
#define CW    60

// workspace layout (bytes)
#define WS_WSUMT   0u          // f16 [64][128]  wsumT[k][c] = sum_o conv_w[o][c][k]
#define WS_LKWT    16384u      // f16 [32][96]   lk_w^T
#define WS_LVWT    22528u      // f16 [112][96]  lv_w^T
#define WS_GATWT   44032u      // f16 [4][32][32] gat_W[h]^T
#define WS_OUTWT   52224u      // f16 [32][128]  out_W^T
#define WS_BSUM    60416u      // float
#define WS_XGAT    60672u      // f16 [B][32][112]  x_gat transposed per sample

// ---- CDNA5 async global<->LDS copies (ASYNCcnt) ----------------------------
__device__ inline void async_ld16(void* lds, const void* glob) {
  asm volatile("global_load_async_to_lds_b128 %0, %1, off"
               :: "v"((unsigned)(uintptr_t)lds), "v"(glob) : "memory");
}
__device__ inline void async_st16(void* glob, const void* lds) {
  asm volatile("global_store_async_from_lds_b128 %0, %1, off"
               :: "v"(glob), "v"((unsigned)(uintptr_t)lds) : "memory");
}
__device__ inline void wait_async() {
  asm volatile("s_wait_asynccnt 0" ::: "memory");
}

// ---- fragment helper (16-bit A/B, contiguous k order: half h holds K = 16h..16h+15) --------------
// lanes 0..15: row (or Bt-row) base+lane, K = k0..k0+15 ; lanes 16..31: K += 16
__device__ inline v16h frag_ld(const h16* base, int row0, int k0, int ld) {
  const int lane = threadIdx.x & 31;
  const int r = row0 + (lane & 15);
  const int k = k0 + ((lane >> 4) << 4);
  return *(const v16h*)(base + r * ld + k);
}

__device__ inline float wred_max(float v) {
#pragma unroll
  for (int s = 16; s > 0; s >>= 1) v = fmaxf(v, __shfl_xor(v, s, 32));
  return v;
}
__device__ inline float wred_sum(float v) {
#pragma unroll
  for (int s = 16; s > 0; s >>= 1) v += __shfl_xor(v, s, 32);
  return v;
}
__device__ inline unsigned wred_or(unsigned v) {
#pragma unroll
  for (int s = 16; s > 0; s >>= 1) v |= __shfl_xor(v, s, 32);
  return v;
}
__device__ inline float eluf(float x)  { return x > 0.f ? x : (expf(x) - 1.f); }
__device__ inline float lrelu(float x) { return x > 0.f ? x : LRA * x; }

// ======================= prep: pack/transpose f16 weights ===================
__global__ __launch_bounds__(256)
void prep_kernel(const float* __restrict__ lk_w, const float* __restrict__ lv_w,
                 const float* __restrict__ gat_W, const float* __restrict__ out_W,
                 const float* __restrict__ conv_w, const float* __restrict__ conv_b,
                 unsigned char* __restrict__ ws) {
  const int tid = threadIdx.x;
  h16* wsum = (h16*)(ws + WS_WSUMT);
  for (int i = tid; i < 64 * 128; i += 256) {
    int k = i >> 7, c = i & 127;
    float v = 0.f;
    if (k < CW && c < NN)
      for (int o = 0; o < NH; ++o) v += conv_w[(o * NN + c) * CW + k];
    wsum[i] = (h16)v;
  }
  h16* lkw = (h16*)(ws + WS_LKWT);
  for (int i = tid; i < 32 * 96; i += 256) {
    int n = i / 96, k = i % 96;
    lkw[i] = (h16)((n < F && k < KD) ? lk_w[k * F + n] : 0.f);
  }
  h16* lvw = (h16*)(ws + WS_LVWT);
  for (int i = tid; i < NP * KDP; i += 256) {
    int n = i / KDP, k = i % KDP;
    lvw[i] = (h16)((n < NN && k < VD) ? lv_w[k * NN + n] : 0.f);
  }
  h16* gw = (h16*)(ws + WS_GATWT);
  for (int i = tid; i < NH * 32 * 32; i += 256) {
    int h = i >> 10, rem = i & 1023, o = rem >> 5, k = rem & 31;
    gw[i] = (h16)((o < F && k < F) ? gat_W[h * F * F + k * F + o] : 0.f);
  }
  h16* ow = (h16*)(ws + WS_OUTWT);
  for (int i = tid; i < 32 * 128; i += 256) {
    int o = i >> 7, k = i & 127;
    ow[i] = (h16)((o < F && k < HF) ? out_W[k * F + o] : 0.f);
  }
  if (tid == 0)
    *(float*)(ws + WS_BSUM) = conv_b[0] + conv_b[1] + conv_b[2] + conv_b[3];
}

// ======================= GAT per sample =====================================
__global__ __launch_bounds__(256)
void gat_kernel(const float* __restrict__ Kmer, const float* __restrict__ pair,
                const float* __restrict__ lk_b,
                const float* __restrict__ ga1, const float* __restrict__ ga2,
                const float* __restrict__ oa1, const float* __restrict__ oa2,
                const float* __restrict__ bng, const float* __restrict__ bnb,
                unsigned char* __restrict__ ws) {
  alignas(64) __shared__ unsigned char smem[63488];
  h16*      HCAT = (h16*)(smem);            // [112][128] (phase-0 A stage; later H2FT f32 [32][112])
  h16*      ATT  = (h16*)(smem + 28672);    // [32][128]  attention chunk (later x_gat staging)
  h16*      XB   = (h16*)(smem + 36864);    // [112][32]  x_kmer (f16)
  h16*      BT0  = (h16*)(smem + 44032);    // [32][128]  weights (Bt)
  h16*      BT1  = (h16*)(smem + 52224);    // [32][128]  Wh^T / Wh2^T (Bt for next GEMM)
  unsigned* MSK  = (unsigned*)(smem + 60416); // [112][4] adjacency bitmask
  float*    F1   = (float*)(smem + 62208);  // [128]
  float*    F2   = (float*)(smem + 62720);  // [128]
  float*    GA   = (float*)(smem + 63232);  // [64]  a1/a2 of current layer

  const int tid = threadIdx.x, wave = tid >> 5, lane = tid & 31;
  const int b = blockIdx.x;
  const float* pb = pair + (size_t)b * NN * NN;

  // prefetch the adjacency block while we stage Kmer and run GEMM0
  for (int off = tid * 128; off < NN * NN * 4; off += 256 * 128)
    __builtin_prefetch((const char*)pb + off, 0, 3);

  // ---- phase 0: x = Kmer @ lk_w + b  (101x84x30), adjacency bitmask --------
  // async: lk_w^T (ws, rows of 192B) -> BT0 (rows of 256B)
  const unsigned char* wlk = ws + WS_LKWT;
  for (int i = tid; i < 32 * 12; i += 256) {
    int n = i / 12, o = (i % 12) << 4;
    async_ld16((unsigned char*)BT0 + n * 256 + o, wlk + n * 192 + o);
  }
  h16* AST = HCAT;  // reuse HCAT region as A stage [112][96]
  for (int i = tid; i < NP * KDP; i += 256) {
    int r = i / KDP, c = i % KDP;
    AST[i] = (h16)((r < NN && c < KD) ? Kmer[(size_t)b * NN * KD + r * KD + c] : 0.f);
  }
  // adjacency bitmask: MSK[r][w] bit m&31 set iff pair[r][m] > 0
  for (int r = wave; r < NP; r += 8) {
#pragma unroll
    for (int q = 0; q < 4; ++q) {
      int m = q * 32 + lane;
      bool adj = (r < NN && m < NN) ? (pb[r * NN + m] > 0.f) : false;
      unsigned w = wred_or(adj ? (1u << lane) : 0u);
      if (lane == 0) MSK[r * 4 + q] = w;
    }
  }
  wait_async();
  __syncthreads();
  for (int job = wave; job < 14; job += 8) {
    int mt = job >> 1, nt = job & 1;
    v8f acc = {};
    for (int ks = 0; ks < 3; ++ks) {
      v16h a  = frag_ld(AST, mt * 16, ks * 32, KDP);
      v16h bb = frag_ld(BT0, nt * 16, ks * 32, 128);
      acc = __builtin_amdgcn_wmma_f32_16x16x32_f16(false, a, false, bb, (short)0, acc, false, false);
    }
    int n = nt * 16 + (lane & 15);
    float bias = (n < F) ? lk_b[n] : 0.f;
#pragma unroll
    for (int v = 0; v < 8; ++v) {
      int m = mt * 16 + v + ((lane >> 4) << 3);
      XB[m * FP + n] = (h16)((n < F) ? (acc[v] + bias) : 0.f);
    }
  }
  __syncthreads();
  for (int i = tid; i < NP * HFP; i += 256) HCAT[i] = (h16)0.f;   // zero hcat
  for (int i = tid; i < 32 * 128; i += 256) BT1[i] = (h16)0.f;    // zero Bt scratch
  __syncthreads();

  // ---- 4 attention heads ---------------------------------------------------
  for (int h = 0; h < NH; ++h) {
    // async: gat_W[h]^T (rows of 64B) -> BT0 (rows of 256B)
    const unsigned char* wg = ws + WS_GATWT + (unsigned)h * 2048u;
    for (int i = tid; i < 32 * 4; i += 256) {
      int n = i >> 2, o = (i & 3) << 4;
      async_ld16((unsigned char*)BT0 + n * 256 + o, wg + n * 64 + o);
    }
    if (tid < F) { GA[tid] = ga1[h * F + tid]; GA[32 + tid] = ga2[h * F + tid]; }
    wait_async();
    __syncthreads();
    // Wh = x @ W_h, stored transposed into BT1 (becomes B of attn GEMM)
    for (int job = wave; job < 14; job += 8) {
      int mt = job >> 1, nt = job & 1;
      v16h a  = frag_ld(XB, mt * 16, 0, FP);
      v16h bb = frag_ld(BT0, nt * 16, 0, 128);
      v8f acc = {};
      acc = __builtin_amdgcn_wmma_f32_16x16x32_f16(false, a, false, bb, (short)0, acc, false, false);
      int n = nt * 16 + (lane & 15);
      int mbase = mt * 16 + ((lane >> 4) << 3);
      v8h pk;
#pragma unroll
      for (int v = 0; v < 8; ++v) pk[v] = (h16)acc[v];
      *(v8h*)(BT1 + n * 128 + mbase) = pk;   // contiguous 16B store
    }
    __syncthreads();
    if (tid < NP) {
      float s1 = 0.f, s2 = 0.f;
      if (tid < NN)
        for (int o = 0; o < F; ++o) {
          float w = (float)BT1[o * 128 + tid];
          s1 += w * GA[o];
          s2 += w * GA[32 + o];
        }
      F1[tid] = s1; F2[tid] = s2;
    }
    __syncthreads();
    // masked softmax + attn @ Wh, in 32-row chunks (lane owns 4 consecutive cols)
    for (int c0 = 0; c0 < NP; c0 += 32) {
      int rows = (NP - c0 < 32) ? (NP - c0) : 32;
      for (int r = c0 + wave; r < c0 + rows; r += 8) {
        unsigned mw = MSK[r * 4 + (lane >> 3)];
        int bitbase = (lane & 7) * 4;
        int mbase = lane * 4;
        float vals[4];
        float rmax = NEGV;
#pragma unroll
        for (int q = 0; q < 4; ++q) {
          float vv = NEGV;
          if (r < NN && ((mw >> (bitbase + q)) & 1u))
            vv = lrelu(F1[r] + F2[mbase + q]);
          vals[q] = vv;
          rmax = fmaxf(rmax, vv);
        }
        rmax = wred_max(rmax);
        float ex[4], rsum = 0.f;
#pragma unroll
        for (int q = 0; q < 4; ++q) { ex[q] = expf(vals[q] - rmax); rsum += ex[q]; }
        rsum = wred_sum(rsum);
        float rinv = 1.f / rsum;
        v4h pk;
#pragma unroll
        for (int q = 0; q < 4; ++q) pk[q] = (h16)((r < NN) ? ex[q] * rinv : 0.f);
        *(v4h*)(ATT + (r - c0) * 128 + mbase) = pk;
      }
      __syncthreads();
      int jobs = (rows >> 4) * 2;
      if (wave < jobs) {
        int mt = wave >> 1, nt = wave & 1;
        v8f acc = {};
        for (int ks = 0; ks < 4; ++ks) {
          v16h a  = frag_ld(ATT, mt * 16, ks * 32, 128);
          v16h bb = frag_ld(BT1, nt * 16, ks * 32, 128);
          acc = __builtin_amdgcn_wmma_f32_16x16x32_f16(false, a, false, bb, (short)0, acc, false, false);
        }
        int nl = nt * 16 + (lane & 15);
        if (nl < F) {
#pragma unroll
          for (int v = 0; v < 8; ++v) {
            int m = c0 + mt * 16 + v + ((lane >> 4) << 3);
            HCAT[m * HFP + h * F + nl] = (h16)eluf(acc[v]);
          }
        }
      }
      __syncthreads();
    }
  }

  // ---- output attention layer ---------------------------------------------
  const unsigned char* wo = ws + WS_OUTWT;
  for (int i = tid; i < 512; i += 256) async_ld16((unsigned char*)BT0 + i * 16, wo + i * 16);
  if (tid < F) { GA[tid] = oa1[tid]; GA[32 + tid] = oa2[tid]; }
  wait_async();
  __syncthreads();
  for (int job = wave; job < 14; job += 8) {  // Wh2 = hcat @ out_W -> BT1^T
    int mt = job >> 1, nt = job & 1;
    v8f acc = {};
    for (int ks = 0; ks < 4; ++ks) {
      v16h a  = frag_ld(HCAT, mt * 16, ks * 32, HFP);
      v16h bb = frag_ld(BT0, nt * 16, ks * 32, 128);
      acc = __builtin_amdgcn_wmma_f32_16x16x32_f16(false, a, false, bb, (short)0, acc, false, false);
    }
    int n = nt * 16 + (lane & 15);
    int mbase = mt * 16 + ((lane >> 4) << 3);
    v8h pk;
#pragma unroll
    for (int v = 0; v < 8; ++v) pk[v] = (h16)acc[v];
    *(v8h*)(BT1 + n * 128 + mbase) = pk;
  }
  __syncthreads();
  if (tid < NP) {
    float s1 = 0.f, s2 = 0.f;
    if (tid < NN)
      for (int o = 0; o < F; ++o) {
        float w = (float)BT1[o * 128 + tid];
        s1 += w * GA[o];
        s2 += w * GA[32 + o];
      }
    F1[tid] = s1; F2[tid] = s2;
  }
  __syncthreads();
  float* H2FT = (float*)HCAT;  // f32 [32][112]: H2FT[feat][node] (HCAT dead now)
  for (int c0 = 0; c0 < NP; c0 += 32) {
    int rows = (NP - c0 < 32) ? (NP - c0) : 32;
    for (int r = c0 + wave; r < c0 + rows; r += 8) {
      unsigned mw = MSK[r * 4 + (lane >> 3)];
      int bitbase = (lane & 7) * 4;
      int mbase = lane * 4;
      float vals[4];
      float rmax = NEGV;
#pragma unroll
      for (int q = 0; q < 4; ++q) {
        float vv = NEGV;
        if (r < NN && ((mw >> (bitbase + q)) & 1u))
          vv = lrelu(F1[r] + F2[mbase + q]);
        vals[q] = vv;
        rmax = fmaxf(rmax, vv);
      }
      rmax = wred_max(rmax);
      float ex[4], rsum = 0.f;
#pragma unroll
      for (int q = 0; q < 4; ++q) { ex[q] = expf(vals[q] - rmax); rsum += ex[q]; }
      rsum = wred_sum(rsum);
      float rinv = 1.f / rsum;
      v4h pk;
#pragma unroll
      for (int q = 0; q < 4; ++q) pk[q] = (h16)((r < NN) ? ex[q] * rinv : 0.f);
      *(v4h*)(ATT + (r - c0) * 128 + mbase) = pk;
    }
    __syncthreads();
    int jobs = (rows >> 4) * 2;
    if (wave < jobs) {
      int mt = wave >> 1, nt = wave & 1;
      v8f acc = {};
      for (int ks = 0; ks < 4; ++ks) {
        v16h a  = frag_ld(ATT, mt * 16, ks * 32, 128);
        v16h bb = frag_ld(BT1, nt * 16, ks * 32, 128);
        acc = __builtin_amdgcn_wmma_f32_16x16x32_f16(false, a, false, bb, (short)0, acc, false, false);
      }
      int nl = nt * 16 + (lane & 15);
      int mbase = c0 + mt * 16 + ((lane >> 4) << 3);
      v8f pk;
#pragma unroll
      for (int v = 0; v < 8; ++v) pk[v] = eluf(acc[v]);
      *(v8f*)(H2FT + nl * NP + mbase) = pk;  // contiguous 32B store, transposed
    }
    __syncthreads();
  }
  // log_softmax over the 30 features of each node
  for (int r = wave; r < NN; r += 8) {
    float v = (lane < F) ? H2FT[lane * NP + r] : NEGV;
    float mx = wred_max(v);
    float ex = (lane < F) ? expf(v - mx) : 0.f;
    float sm = wred_sum(ex);
    if (lane < F) H2FT[lane * NP + r] = v - mx - logf(sm);
  }
  __syncthreads();
  // per-sample BatchNorm over the 101 nodes (contiguous per feature now)
  if (tid < F) {
    float mean = 0.f;
    for (int r = 0; r < NN; ++r) mean += H2FT[tid * NP + r];
    mean *= (1.f / NN);
    float var = 0.f;
    for (int r = 0; r < NN; ++r) { float d = H2FT[tid * NP + r] - mean; var += d * d; }
    var *= (1.f / NN);
    F1[tid] = mean;
    F2[tid] = bng[tid] * rsqrtf(var + EPSBN);
  }
  __syncthreads();
  // stage x_gat (feature-major) into LDS, then bulk async store LDS->global
  h16* XGS = ATT;  // 7168B staging (ATT is free now)
  for (int i = tid; i < 32 * NP; i += 256) {
    int t = i / NP, c = i % NP;
    float v = 0.f;
    if (t < F && c < NN) v = fmaxf((H2FT[i] - F1[t]) * F2[t] + bnb[t], 0.f);
    XGS[i] = (h16)v;
  }
  __syncthreads();
  unsigned char* xg = ws + WS_XGAT + (size_t)b * (32 * NP * 2);
  for (int i = tid; i < 448; i += 256)
    async_st16(xg + i * 16, (const unsigned char*)XGS + i * 16);
  // S_ENDPGM performs an implicit wait-idle (covers ASYNCcnt)
}

// ======================= vec projection + band conv per sample ==============
__global__ __launch_bounds__(256)
void conv_kernel(const float* __restrict__ vec, const float* __restrict__ lv_b,
                 const unsigned char* __restrict__ ws, float* __restrict__ out) {
  alignas(64) __shared__ unsigned char smem[65024];
  h16*   XT   = (h16*)(smem);                  // [144][128]  X^T (feat x node)
  h16*   REG  = (h16*)(smem + 36864);          // phase A: lv_w^T [112][96]; phase B: wsumT [64][128]
  h16*   ACH  = (h16*)(smem + 36864 + 21504);  // [32][96]  vec^T chunk
  float* PCH  = (float*)(smem + 36864 + 16384);// [64][32]  P chunk, transposed (phase B)
  float* OUTA = (float*)(smem + 64512);        // [128] accumulators

  const int tid = threadIdx.x, wave = tid >> 5, lane = tid & 31;
  const int b = blockIdx.x;

  // async: lv_w^T -> REG (contiguous 21504B)
  const unsigned char* wlv = ws + WS_LVWT;
  for (int i = tid; i < 1344; i += 256) async_ld16((unsigned char*)REG + i * 16, wlv + i * 16);
  // async: x_gat [32][112] (rows of 224B) -> XT rows 0..31 (rows of 256B)
  const unsigned char* xg = ws + WS_XGAT + (size_t)b * (32 * NP * 2);
  for (int i = tid; i < 32 * 14; i += 256) {
    int n = i / 14, o = (i % 14) << 4;
    async_ld16((unsigned char*)XT + n * 256 + o, xg + n * 224 + o);
  }
  // zero only the XT regions not written by the async copy or GEMM1:
  // cols 112..127 (all rows) and rows 131..143 (all cols)
  for (int i = tid; i < 144 * 16; i += 256) {
    int r = i >> 4, c = 112 + (i & 15);
    XT[r * 128 + c] = (h16)0.f;
  }
  for (int i = tid; i < 13 * 128; i += 256) {
    int r = 131 + (i >> 7), c = i & 127;
    XT[r * 128 + c] = (h16)0.f;
  }
  if (tid < 128) OUTA[tid] = 0.f;
  wait_async();
  __syncthreads();

  // GEMM1: Xvec^T = vec^T @ lv_w  (rows 30..130 of XT), 4 M-chunks of 32
  for (int m0 = 0; m0 < NP; m0 += 32) {
    for (int i = tid; i < 32 * KDP; i += 256) {
      int jl = i / KDP, d = i % KDP;
      int j = m0 + jl;
      ACH[i] = (h16)((j < VL && d < VD) ? vec[(size_t)b * VD * VL + d * VL + j] : 0.f);
    }
    __syncthreads();
    for (int job = wave; job < 14; job += 8) {
      int mt = job / 7, nt = job % 7;
      v8f acc = {};
      for (int ks = 0; ks < 3; ++ks) {
        v16h a  = frag_ld(ACH, mt * 16, ks * 32, KDP);
        v16h bb = frag_ld(REG, nt * 16, ks * 32, KDP);
        acc = __builtin_amdgcn_wmma_f32_16x16x32_f16(false, a, false, bb, (short)0, acc, false, false);
      }
      int c = nt * 16 + (lane & 15);
      float bias = (c < NN) ? lv_b[c] : 0.f;
#pragma unroll
      for (int v = 0; v < 8; ++v) {
        int j = m0 + mt * 16 + v + ((lane >> 4) << 3);
        int row = F + j;
        if (row < 144) XT[row * 128 + c] = (h16)((j < VL && c < NN) ? acc[v] + bias : 0.f);
      }
    }
    __syncthreads();
  }

  // GEMM2: P = X^T @ wsum, banded reduction out[t] = sum_k P[t+k][k]
  const unsigned char* wws = ws + WS_WSUMT;
  for (int i = tid; i < 1024; i += 256) async_ld16((unsigned char*)REG + i * 16, wws + i * 16);
  wait_async();
  __syncthreads();
  for (int m0 = 0; m0 < 144; m0 += 32) {
    int mtiles = (144 - m0 >= 32) ? 2 : 1;
    if (wave < mtiles * 4) {
      int mt = wave >> 2, nt = wave & 3;
      v8f acc = {};
      for (int ks = 0; ks < 4; ++ks) {
        v16h a  = frag_ld(XT, m0 + mt * 16, ks * 32, 128);
        v16h bb = frag_ld(REG, nt * 16, ks * 32, 128);
        acc = __builtin_amdgcn_wmma_f32_16x16x32_f16(false, a, false, bb, (short)0, acc, false, false);
      }
      int col = nt * 16 + (lane & 15);
      int rbase = mt * 16 + ((lane >> 4) << 3);
      *(v8f*)(PCH + col * 32 + rbase) = acc;   // contiguous 32B store, transposed
    }
    __syncthreads();
    for (int i = tid; i < 64 * 32; i += 256) {
      int k = i >> 5, rl = i & 31;
      int r = m0 + rl;
      if (rl < mtiles * 16 && r < 144 && k < CW) {
        int t = r - k;
        if (t >= 0 && t < TOUT) atomicAdd(&OUTA[t], PCH[i]);
      }
    }
    __syncthreads();
  }
  float bsum = *(const float*)(ws + WS_BSUM);
  if (tid < TOUT) {
    float v = OUTA[tid] + bsum;
    out[(size_t)b * TOUT + tid] = 1.f / (1.f + expf(-v));
  }
}

// ======================= launch =============================================
extern "C" void kernel_launch(void* const* d_in, const int* in_sizes, int n_in,
                              void* d_out, int out_size, void* d_ws, size_t ws_size,
                              hipStream_t stream) {
  (void)in_sizes; (void)n_in; (void)out_size; (void)ws_size;
  const float* Kmer   = (const float*)d_in[0];
  const float* vec    = (const float*)d_in[1];
  const float* pair   = (const float*)d_in[2];
  const float* lk_w   = (const float*)d_in[4];
  const float* lk_b   = (const float*)d_in[5];
  const float* lv_w   = (const float*)d_in[6];
  const float* lv_b   = (const float*)d_in[7];
  const float* gat_W  = (const float*)d_in[8];
  const float* ga1    = (const float*)d_in[9];
  const float* ga2    = (const float*)d_in[10];
  const float* out_W  = (const float*)d_in[11];
  const float* oa1    = (const float*)d_in[12];
  const float* oa2    = (const float*)d_in[13];
  const float* bng    = (const float*)d_in[14];
  const float* bnb    = (const float*)d_in[15];
  const float* conv_w = (const float*)d_in[16];
  const float* conv_b = (const float*)d_in[17];
  unsigned char* ws   = (unsigned char*)d_ws;
  float* out          = (float*)d_out;

  prep_kernel<<<1, 256, 0, stream>>>(lk_w, lv_w, gat_W, out_W, conv_w, conv_b, ws);
  gat_kernel<<<BATCH, 256, 0, stream>>>(Kmer, pair, lk_b, ga1, ga2, oa1, oa2, bng, bnb, ws);
  conv_kernel<<<BATCH, 256, 0, stream>>>(vec, lv_b, ws, out);
}
